// TDRLoss_15788299780582
// MI455X (gfx1250) — compile-verified, matrix operand order checked by reading, not run on hardware
//
#include <hip/hip_runtime.h>
#include <hip/hip_bf16.h>

// ---------------------------------------------------------------------------
// DETR-style matcher + focal loss for MI455X (gfx1250, wave32, WMMA).
//
// Pipeline:
//  K1 cost_kernel : cost[b][g][q] = -2*sigmoid(cls)[q,label[g]] + 0.25*L1(q,g)
//                   class-gather done as GEMM vs one-hot via
//                   v_wmma_f32_16x16x32_f16 (K=128 in 4 steps, 4 g-tiles),
//                   B fragments materialized ahead of the 4 back-to-back
//                   WMMAs to avoid WMMA->VALU WAR hazard NOPs.
//  K2 lsa_kernel  : Jonker-Volgenant shortest augmenting path, one WGP/batch,
//                   all dual/solver state in LDS, Q-wide parallel min-scan
//                   (cost matrix stays resident in the 192 MB L2).
//  K3 focal_neg   : streaming all-negative focal sum over B*Q*C. float4
//                   loads + v_rcp/fast-log math keep it HBM-bound
//                   (64 MB @ 23.3 TB/s ~ 3us).
//  K4 matched     : per-(b,g) corrections (pos-vs-neg focal delta, bbox L1,
//                   xyz err, argmax accuracy).
//  K5 finalize    : 5 output scalars.
// ---------------------------------------------------------------------------

typedef __attribute__((ext_vector_type(16))) _Float16 v16h;
typedef __attribute__((ext_vector_type(8)))  float    v8f;

#define B_  64
#define Q_  2048
#define G_  64
#define C_  128
#define INF_ 1e30f

__constant__ float c_norm[10] = {1.f, 1.f, 1.f, 0.1f, 0.1f, 0.1f, 1.f, 1.f, 0.1f, 0.1f};

// Fast sigmoid via single v_rcp_f32 (no IEEE div expansion).
__device__ __forceinline__ float sigmoid_fast(float x) {
    float t = __expf(-fabsf(x));
    float r = __builtin_amdgcn_rcpf(1.0f + t);
    return (x >= 0.f) ? r : t * r;
}
// Fused sigmoid + softplus sharing one exp: p = sigma(x), sp = log(1+e^x).
__device__ __forceinline__ void sig_softplus_fast(float x, float& p, float& sp) {
    float t = __expf(-fabsf(x));
    float r = __builtin_amdgcn_rcpf(1.0f + t);
    p  = (x >= 0.f) ? r : t * r;
    sp = fmaxf(x, 0.f) + __logf(1.0f + t);
}

// ---------------------------------------------------------------------------
// K1: cost matrix. Grid (Q/64, B), block 128 (4 waves); each wave owns a
// 16-query tile, computes 16x64 cost_cls via 16 WMMA ops, adds L1 term.
// ---------------------------------------------------------------------------
__global__ __launch_bounds__(128)
void cost_kernel(const float* __restrict__ cls,
                 const float* __restrict__ bbox,
                 const int*   __restrict__ glab,
                 const float* __restrict__ gbox,
                 float* __restrict__ cost)
{
    const int b   = blockIdx.y;
    const int q0  = blockIdx.x * 64;       // 64 queries per block
    const int tid = threadIdx.x;           // 0..127
    const int wv  = tid >> 5;              // wave 0..3
    const int ln  = tid & 31;              // lane in wave

    __shared__ float gn_s[G_][10];
    __shared__ float pn_s[64][10];
    __shared__ int   lab_s[G_];

    for (int i = tid; i < G_; i += 128) lab_s[i] = glab[b * G_ + i];
    for (int i = tid; i < G_ * 10; i += 128) {
        int g = i / 10, d = i % 10;
        float v = gbox[((size_t)b * G_ + g) * 10 + d] / c_norm[d];
        gn_s[g][d] = fminf(fmaxf(v, -100.f), 100.f);
    }
    for (int i = tid; i < 64 * 10; i += 128) {
        int qq = i / 10, d = i % 10;
        float v = bbox[((size_t)b * Q_ + q0 + qq) * 10 + d] / c_norm[d];
        pn_s[qq][d] = fminf(fmaxf(v, -100.f), 100.f);
    }
    __syncthreads();

    const int m  = ln & 15;                // A-matrix row held by this lane
    const int kg = ln >> 4;                // K-half select (ISA A layout)
    const int q  = q0 + wv * 16 + m;
    const float* row = cls + ((size_t)b * Q_ + q) * C_;

    // one-hot labels for this lane's B column, one per g-tile (hoisted)
    const int lab0 = lab_s[ 0 + (ln & 15)];
    const int lab1 = lab_s[16 + (ln & 15)];
    const int lab2 = lab_s[32 + (ln & 15)];
    const int lab3 = lab_s[48 + (ln & 15)];

    v8f acc0 = {}, acc1 = {}, acc2 = {}, acc3 = {};

    #pragma unroll
    for (int kk = 0; kk < 4; ++kk) {       // K=128 in chunks of 32
        const int cbase = kk * 32;
        // A fragment: halves 0..7 -> K=kg*8+h ; 8..15 -> K=16+kg*8+(h-8)
        const float4* p0 = (const float4*)(row + cbase + kg * 8);
        const float4* p1 = (const float4*)(row + cbase + 16 + kg * 8);
        float4 f0 = p0[0], f1 = p0[1], f2 = p1[0], f3 = p1[1];
        float xs[16] = {f0.x, f0.y, f0.z, f0.w, f1.x, f1.y, f1.z, f1.w,
                        f2.x, f2.y, f2.z, f2.w, f3.x, f3.y, f3.z, f3.w};
        v16h a;
        #pragma unroll
        for (int h = 0; h < 16; ++h) a[h] = (_Float16)sigmoid_fast(xs[h]);

        if (kk < 3) __builtin_prefetch(row + cbase + 32, 0, 1); // global_prefetch_b8

        // Materialize ALL four one-hot B fragments first (distinct registers),
        // then issue the 4 WMMAs back-to-back (no D->A/B RAW between them).
        v16h b0, b1, b2, b3;
        #pragma unroll
        for (int h = 0; h < 16; ++h) {
            int K = cbase + ((h < 8) ? (kg * 8 + h) : (16 + kg * 8 + (h - 8)));
            b0[h] = (_Float16)((lab0 == K) ? 1.0f : 0.0f);
            b1[h] = (_Float16)((lab1 == K) ? 1.0f : 0.0f);
            b2[h] = (_Float16)((lab2 == K) ? 1.0f : 0.0f);
            b3[h] = (_Float16)((lab3 == K) ? 1.0f : 0.0f);
        }
        acc0 = __builtin_amdgcn_wmma_f32_16x16x32_f16(false, a, false, b0, (short)0, acc0, false, false);
        acc1 = __builtin_amdgcn_wmma_f32_16x16x32_f16(false, a, false, b1, (short)0, acc1, false, false);
        acc2 = __builtin_amdgcn_wmma_f32_16x16x32_f16(false, a, false, b2, (short)0, acc2, false, false);
        acc3 = __builtin_amdgcn_wmma_f32_16x16x32_f16(false, a, false, b3, (short)0, acc3, false, false);
    }

    // C/D layout: vgpr r, lanes 0-15 -> row r, lanes 16-31 -> row r+8; col = ln&15
    v8f accs[4] = {acc0, acc1, acc2, acc3};
    #pragma unroll
    for (int gt = 0; gt < 4; ++gt) {
        const int g = gt * 16 + (ln & 15);
        #pragma unroll
        for (int r = 0; r < 8; ++r) {
            int ml = (ln < 16) ? r : (r + 8);
            int ql = wv * 16 + ml;
            float l1 = 0.f;
            #pragma unroll
            for (int d = 0; d < 10; ++d) l1 += fabsf(pn_s[ql][d] - gn_s[g][d]);
            float cv = -2.0f * accs[gt][r] + 0.25f * l1;  // COST_CLS*(-prob) + COST_BBOX*L1
            cost[((size_t)b * G_ + g) * Q_ + q0 + ql] = cv;
        }
    }
}

// ---------------------------------------------------------------------------
// K2: Jonker-Volgenant LSA, one workgroup per batch, 64x2048 cost (G rows).
// ---------------------------------------------------------------------------
__global__ __launch_bounds__(256)
void lsa_kernel(const float* __restrict__ cost, int* __restrict__ pred)
{
    const int b   = blockIdx.x;
    const int tid = threadIdx.x;

    __shared__ float v_s[Q_ + 1];
    __shared__ float minv_s[Q_ + 1];
    __shared__ int   way_s[Q_ + 1];
    __shared__ int   p_s[Q_ + 1];
    __shared__ float u_s[G_ + 1];
    __shared__ unsigned used_s[(Q_ + 32) / 32 + 1];
    __shared__ float rval[256];
    __shared__ int   ridx[256];
    __shared__ int   j0_sh;
    __shared__ float delta_sh;
    __shared__ int   done_sh;

    for (int j = tid; j <= Q_; j += 256) { v_s[j] = 0.f; p_s[j] = 0; way_s[j] = 0; }
    for (int i = tid; i <= G_; i += 256) u_s[i] = 0.f;
    __syncthreads();

    const float* cb = cost + (size_t)b * G_ * Q_;

    for (int i = 1; i <= G_; ++i) {
        for (int j = tid; j <= Q_; j += 256) minv_s[j] = INF_;
        for (int w = tid; w < (Q_ + 32) / 32 + 1; w += 256) used_s[w] = 0u;
        if (tid == 0) { p_s[0] = i; j0_sh = 0; done_sh = 0; }
        __syncthreads();

        for (int iter = 0; iter < 4096; ++iter) {
            int j0 = j0_sh;
            if (tid == 0) used_s[j0 >> 5] |= (1u << (j0 & 31));
            __syncthreads();
            int i0 = p_s[j0];
            float ui0 = u_s[i0];
            const float* rowp = cb + (size_t)(i0 - 1) * Q_;

            float best = INF_; int bidx = 1;
            for (int j = 1 + tid; j <= Q_; j += 256) {      // coalesced scan
                if (!((used_s[j >> 5] >> (j & 31)) & 1u)) {
                    float cur = rowp[j - 1] - ui0 - v_s[j];
                    if (cur < minv_s[j]) { minv_s[j] = cur; way_s[j] = j0; }
                    float mv = minv_s[j];
                    if (mv < best) { best = mv; bidx = j; }
                }
            }
            rval[tid] = best; ridx[tid] = bidx;
            __syncthreads();
            for (int s = 128; s > 0; s >>= 1) {
                if (tid < s && rval[tid + s] < rval[tid]) {
                    rval[tid] = rval[tid + s]; ridx[tid] = ridx[tid + s];
                }
                __syncthreads();
            }
            if (tid == 0) { delta_sh = rval[0]; j0_sh = ridx[0]; }
            __syncthreads();
            float delta = delta_sh;
            int j1 = j0_sh;

            for (int j = 1 + tid; j <= Q_; j += 256) {      // dual updates
                if ((used_s[j >> 5] >> (j & 31)) & 1u) {
                    u_s[p_s[j]] += delta;                   // distinct rows per used col
                    v_s[j]      -= delta;
                } else {
                    minv_s[j]   -= delta;
                }
            }
            if (tid == 0) {
                u_s[p_s[0]] += delta;                       // column 0 is always used
                v_s[0]      -= delta;
                if (p_s[j1] == 0) done_sh = 1;
            }
            __syncthreads();
            if (done_sh) break;
        }
        if (tid == 0) {                                     // augment (serial, short)
            int j0 = j0_sh;
            while (j0) { int j1 = way_s[j0]; p_s[j0] = p_s[j1]; j0 = j1; }
        }
        __syncthreads();
    }
    for (int j = 1 + tid; j <= Q_; j += 256) {
        int pi = p_s[j];
        if (pi) pred[b * G_ + pi - 1] = j - 1;
    }
}

// ---------------------------------------------------------------------------
// K3: all-negative focal sum over B*Q*C. float4 loads, fast math: HBM bound.
// Grid sized so every thread's 8 elements are in range (N = 8192*2048).
// ---------------------------------------------------------------------------
__global__ __launch_bounds__(256)
void focal_neg_kernel(const float* __restrict__ cls, float* __restrict__ acc)
{
    __shared__ float red[256];
    const size_t N = (size_t)B_ * Q_ * C_;
    size_t base = ((size_t)blockIdx.x * 256 + threadIdx.x) * 8;
    float s = 0.f;
    if (base + 8 <= N) {
        const float4* v4 = (const float4*)(cls + base);
        float4 c0 = v4[0], c1 = v4[1];
        float xv[8] = {c0.x, c0.y, c0.z, c0.w, c1.x, c1.y, c1.z, c1.w};
        #pragma unroll
        for (int t = 0; t < 8; ++t) {
            float p, sp;
            sig_softplus_fast(xv[t], p, sp);     // sp = -log_sigmoid(-x)
            s += 0.75f * p * p * sp;             // (1-alpha)*p^2*ce_neg
        }
    }
    red[threadIdx.x] = s; __syncthreads();
    for (int st = 128; st > 0; st >>= 1) {
        if (threadIdx.x < st) red[threadIdx.x] += red[threadIdx.x + st];
        __syncthreads();
    }
    if (threadIdx.x == 0) atomicAdd(&acc[0], red[0]);
}

// ---------------------------------------------------------------------------
// K4: matched-pair corrections (4096 pairs)
// ---------------------------------------------------------------------------
__global__ __launch_bounds__(256)
void matched_kernel(const float* __restrict__ cls, const float* __restrict__ bbox,
                    const int* __restrict__ glab, const float* __restrict__ gbox,
                    const int* __restrict__ pred, float* __restrict__ acc)
{
    int idx = blockIdx.x * blockDim.x + threadIdx.x;
    if (idx >= B_ * G_) return;
    int b = idx / G_;
    int q = pred[idx];
    int lab = glab[idx];

    const float* crow = cls + ((size_t)b * Q_ + q) * C_;
    float x = crow[lab];
    float p, sp_pos_arg;
    sig_softplus_fast(x, p, sp_pos_arg);              // sp_pos_arg = softplus(x)
    float sp_neg_x;                                   // softplus(-x)
    {
        float t = __expf(-fabsf(x));
        sp_neg_x = fmaxf(-x, 0.f) + __logf(1.0f + t);
    }
    // swap this element from negative-class to positive-class focal term
    float dlt = 0.25f * (1.f - p) * (1.f - p) * sp_neg_x
              - 0.75f * p * p * sp_pos_arg;
    atomicAdd(&acc[0], dlt);

    const float* mp = bbox + ((size_t)b * Q_ + q) * 10;
    const float* mg = gbox + (size_t)idx * 10;
    float l1 = 0.f, xyz = 0.f;
    #pragma unroll
    for (int d = 0; d < 10; ++d) {
        l1 += fabsf(mp[d] / c_norm[d] - mg[d] / c_norm[d]);
        if (d < 3) xyz += fabsf(mp[d] - mg[d]);
    }
    atomicAdd(&acc[1], l1);
    atomicAdd(&acc[3], xyz);

    int am = 0; float bv = crow[0];
    #pragma unroll 4
    for (int c = 1; c < C_; ++c) {
        float v = crow[c];
        if (v > bv) { bv = v; am = c; }
    }
    atomicAdd(&acc[2], (am == lab) ? 1.f : 0.f);
}

__global__ void zero_acc_kernel(float* acc) {
    if (threadIdx.x < 8) acc[threadIdx.x] = 0.f;
}

__global__ void finalize_kernel(const float* __restrict__ acc, float* __restrict__ out)
{
    if (threadIdx.x == 0) {
        out[0] = acc[0] / (float)(G_ * B_);          // loss_cls
        out[1] = acc[1] / (float)(B_ * G_ * 10);     // loss_bbox (mean)
        out[2] = (float)G_;                          // matched
        out[3] = acc[2] / (float)(B_ * G_);          // pos_acc
        out[4] = acc[3] / (float)(B_ * G_ * 3);      // xyz_err
    }
}

// ---------------------------------------------------------------------------
extern "C" void kernel_launch(void* const* d_in, const int* in_sizes, int n_in,
                              void* d_out, int out_size, void* d_ws, size_t ws_size,
                              hipStream_t stream)
{
    const float* cls  = (const float*)d_in[0];   // (B,Q,C) f32
    const float* bbox = (const float*)d_in[1];   // (B,Q,10) f32
    const int*   glab = (const int*)d_in[2];     // (B,G) i32
    const float* gbox = (const float*)d_in[3];   // (B,G,10) f32
    float* out = (float*)d_out;

    // workspace layout: cost (32 MB, lives in the 192 MB L2) | pred | acc
    float* cost = (float*)d_ws;
    int*   pred = (int*)(cost + (size_t)B_ * G_ * Q_);
    float* acc  = (float*)(pred + B_ * G_);

    zero_acc_kernel<<<1, 32, 0, stream>>>(acc);

    dim3 g1(Q_ / 64, B_);
    cost_kernel<<<g1, 128, 0, stream>>>(cls, bbox, glab, gbox, cost);

    lsa_kernel<<<B_, 256, 0, stream>>>(cost, pred);

    size_t N = (size_t)B_ * Q_ * C_;
    int nb = (int)(N / 2048);                     // exact: 8192 blocks
    focal_neg_kernel<<<nb, 256, 0, stream>>>(cls, acc);

    matched_kernel<<<(B_ * G_ + 255) / 256, 256, 0, stream>>>(cls, bbox, glab, gbox, pred, acc);

    finalize_kernel<<<1, 32, 0, stream>>>(acc, out);
}
